// LeNet1_6528350290521
// MI455X (gfx1250) — matrix operand order repaired and checked by reading, not compiled, not run on hardware
//
#include <hip/hip_runtime.h>
#include <hip/hip_bf16.h>

typedef __attribute__((ext_vector_type(16))) _Float16     v16h;
typedef __attribute__((ext_vector_type(8)))  float        v8f;
typedef __attribute__((ext_vector_type(8)))  unsigned int v8u;
typedef __attribute__((ext_vector_type(4)))  _Float16     h4v;
typedef __attribute__((ext_vector_type(2)))  _Float16     h2v;

// ScaledTanh 1.7159*tanh(2x/3) = A - 2A/(exp2(c*x)+1), c = (4/3)*log2(e).
// Uses v_exp_f32 (exp2) + v_rcp_f32 explicitly: 3 VALU + 2 trans ops.
__device__ __forceinline__ float stanh_(float x) {
    float e = __builtin_amdgcn_exp2f(1.9235933878519513f * x);
    float r = __builtin_amdgcn_rcpf(e + 1.0f);
    return fmaf(-3.4318f, r, 1.7159f);
}

// LeNet-1 C3 connection table, bit (outCh*4 + inCh)
__device__ __forceinline__ bool conn_(int outc, int inc) {
    const unsigned long long M =
        (1ull << 0)  | (3ull << 4)  | (3ull << 8)  | (1ull << 13) |
        (3ull << 16) | (3ull << 20) | (1ull << 26) | (3ull << 30) |
        (3ull << 34) | (1ull << 39) | (3ull << 42) | (3ull << 46);
    return (M >> (outc * 4 + inc)) & 1ull;
}

// One wave == one image, 4 images per wave; block = 8 waves = 32 images.
__global__ __launch_bounds__(256) void lenet1_fused(
    const float* __restrict__ x,
    const float* __restrict__ h1_w, const float* __restrict__ h1_b,
    const float* __restrict__ p2_w, const float* __restrict__ p2_b,
    const float* __restrict__ c3_w, const float* __restrict__ c3_b,
    const float* __restrict__ p4_w, const float* __restrict__ p4_b,
    float* __restrict__ out, int Btot)
{
    // input image, f16
    __shared__ __align__(16) _Float16 s_xh[8][784];
    // H2 activations [4ch][12row][16col-padded] + overflow pad, two copies:
    // even copy (value(c) at slot c) and odd copy (value(c) at slot c+1) so
    // every f16 pair load is a 4B-aligned ds_load_b32 for any pixel parity.
    __shared__ __align__(16) _Float16 s_h2e[8][840];
    __shared__ __align__(16) _Float16 s_h2o[8][840];

    const int tid  = threadIdx.x;
    const int wv   = tid >> 5;
    const int lane = tid & 31;

    // ---- zero H2 arrays once: pad slots must be finite (garbage*0 in WMMA
    // must not be NaN); data slots are rewritten each image. -----------------
    {
        unsigned int* ze = (unsigned int*)&s_h2e[wv][0];
        unsigned int* zo = (unsigned int*)&s_h2o[wv][0];
        #pragma unroll
        for (int i = 0; i < 14; ++i) {
            int idx = lane + 32 * i;                 // 420 dwords per array
            if (idx < 420) { ze[idx] = 0u; zo[idx] = 0u; }
        }
    }

    // ---------------- H1/H2 per-lane setup (channel chh, 8 lanes each) -----
    const int chh = lane >> 3;
    const int sj  = lane & 7;
    h2v w2r[25];                                     // {w,w} duplicated halves
    #pragma unroll
    for (int i = 0; i < 25; ++i) {
        _Float16 wh = (_Float16)h1_w[chh * 25 + i];
        h2v t; t.x = wh; t.y = wh; w2r[i] = t;
    }
    const _Float16 b1h = (_Float16)h1_b[chh];
    h2v bias1; bias1.x = b1h; bias1.y = b1h;
    const float pw2 = p2_w[chh];
    const float pb2 = p2_b[chh];

    // ---------------- H3/H4 per-lane setup ---------------------------------
    const int chc = lane & 15;                       // output column/channel
    const int hi  = lane >> 4;                       // K half
    const float bias3 = (chc < 12) ? c3_b[chc] : 0.f;
    const float pw4   = (chc < 12) ? p4_w[chc] : 0.f;
    const float pb4   = (chc < 12) ? p4_b[chc] : 0.f;

    // B fragments: K block s = input channel, padded tap map t->(t/6, t%6);
    // slots with dy==5 or dx==5 (and unconnected pairs) are zero so the
    // matching garbage A elements are annihilated.
    v16h bfr[4];
    #pragma unroll
    for (int s = 0; s < 4; ++s) {
        #pragma unroll
        for (int e = 0; e < 16; ++e) {
            int t = (e < 8) ? (hi * 8 + e) : (16 + hi * 8 + (e - 8)); // tap slot 0..31
            int dy = t / 6, dx = t % 6;
            float v = 0.f;
            if (dy < 5 && dx < 5 && chc < 12 && conn_(chc, s))
                v = c3_w[(chc * 4 + s) * 25 + dy * 5 + dx];
            bfr[s][e] = (_Float16)v;
        }
    }

    // A-side geometry: row M = lane%16 -> pixel (py = 2*tile + m/8, px = m%8)
    const int m   = lane & 15;
    const int py0 = m >> 3;
    const int px  = m & 7;
    const int pxh = (px + (px & 1)) >> 1;            // pair index incl. parity
    const unsigned int* hbase = (px & 1)
        ? (const unsigned int*)&s_h2o[wv][0]
        : (const unsigned int*)&s_h2e[wv][0];
    // per-fragment-VGPR tap offsets (depend on lane half hi)
    int woff[8];
    #pragma unroll
    for (int jv = 0; jv < 8; ++jv) {
        int t = (jv < 4) ? (hi * 8 + 2 * jv) : (16 + hi * 8 + 2 * (jv - 4));
        woff[jv] = (t / 6) * 8 + (t % 6) / 2;        // dy*8 + dx/2 (dwords)
    }

    for (int it = 0; it < 4; ++it) {
        const int img = (blockIdx.x * 8 + wv) * 4 + it;   // wave-uniform
        if (img >= Btot) break;

        // ---------------- phase 1: stage input image as f16 ----------------
        {
            const float4* xs = (const float4*)(x + (size_t)img * 784);
            #pragma unroll
            for (int i = 0; i < 7; ++i) {
                int idx = lane + 32 * i;             // 196 float4 per image
                if (idx < 196) {
                    float4 v4 = xs[idx];
                    h4v h4;
                    h4.x = (_Float16)v4.x; h4.y = (_Float16)v4.y;
                    h4.z = (_Float16)v4.z; h4.w = (_Float16)v4.w;
                    *(h4v*)&s_xh[wv][idx * 4] = h4;
                }
            }
        }

        // ------- phase 2: H1 conv (packed f16) + stanh + H2 sum-pool -------
        // 576 H2 cells = 4ch*12*12; lane -> (chh, sj), 18 cells each.
        const unsigned int* xw = (const unsigned int*)&s_xh[wv][0];
        for (int i2 = 0; i2 < 18; ++i2) {
            int j = sj + 8 * i2;                     // 0..143
            int r = j / 12, cx = j % 12;
            h2v acc0 = bias1;                        // subpixels sy=0, sx=0/1
            h2v acc1 = bias1;                        // subpixels sy=1, sx=0/1
            #pragma unroll
            for (int y = 0; y < 6; ++y) {
                int rb = (2 * r + y) * 14 + cx;      // dword index
                unsigned int u0 = xw[rb + 0];
                unsigned int u1 = xw[rb + 1];
                unsigned int u2 = xw[rb + 2];
                h2v P0 = __builtin_bit_cast(h2v, u0);
                h2v P1 = __builtin_bit_cast(h2v, u1);
                h2v P2 = __builtin_bit_cast(h2v, u2);
                h2v S1; S1.x = P0.y; S1.y = P1.x;    // shifted pairs
                h2v S3; S3.x = P1.y; S3.y = P2.x;
                h2v S[5] = { P0, S1, P1, S3, P2 };
                if (y < 5) {
                    #pragma unroll
                    for (int d = 0; d < 5; ++d)
                        acc0 = __builtin_elementwise_fma(w2r[y * 5 + d], S[d], acc0);
                }
                if (y > 0) {
                    #pragma unroll
                    for (int d = 0; d < 5; ++d)
                        acc1 = __builtin_elementwise_fma(w2r[(y - 1) * 5 + d], S[d], acc1);
                }
            }
            float pool = stanh_((float)acc0.x) + stanh_((float)acc0.y)
                       + stanh_((float)acc1.x) + stanh_((float)acc1.y);
            _Float16 hv = (_Float16)stanh_(fmaf(pw2, pool, pb2));
            int hidx = chh * 192 + r * 16 + cx;
            s_h2e[wv][hidx]     = hv;                // aligned copy
            s_h2o[wv][hidx + 1] = hv;                // shifted copy
        }

        // ---------------- phase 3+4: C3 WMMA + fused H4 --------------------
        #pragma unroll
        for (int t = 0; t < 4; ++t) {
            v8f acc;
            #pragma unroll
            for (int jj = 0; jj < 8; ++jj) acc[jj] = bias3;

            const int py = 2 * t + py0;
            const unsigned int* hb2 = hbase + py * 8 + pxh;
            const unsigned int* pj[8];
            #pragma unroll
            for (int jv = 0; jv < 8; ++jv) pj[jv] = hb2 + woff[jv];

            #pragma unroll
            for (int s = 0; s < 4; ++s) {
                v8u au;
                #pragma unroll
                for (int jv = 0; jv < 8; ++jv) au[jv] = pj[jv][s * 96];
                v16h af = __builtin_bit_cast(v16h, au);
                acc = __builtin_amdgcn_wmma_f32_16x16x32_f16(
                        false, af, false, bfr[s], (short)0, acc, false, false);
            }

            // C layout: lane = col chc; VGPR j -> pixel row 2t+hi, col j.
            float h3[8];
            #pragma unroll
            for (int jj = 0; jj < 8; ++jj) h3[jj] = stanh_(acc[jj]);

            #pragma unroll
            for (int q = 0; q < 4; ++q) {
                float p = h3[2 * q] + h3[2 * q + 1];     // horizontal pair
                p += __shfl_xor(p, 16, 32);              // vertical pair
                float ov = stanh_(fmaf(pw4, p, pb4));
                if (lane < 12)                           // chc<12 && hi==0
                    out[(size_t)img * 192 + chc * 16 + t * 4 + q] = ov;
            }
        }
    }
}

extern "C" void kernel_launch(void* const* d_in, const int* in_sizes, int n_in,
                              void* d_out, int out_size, void* d_ws, size_t ws_size,
                              hipStream_t stream) {
    const float* x    = (const float*)d_in[0];
    const float* h1_w = (const float*)d_in[1];
    const float* h1_b = (const float*)d_in[2];
    const float* p2_w = (const float*)d_in[3];
    const float* p2_b = (const float*)d_in[4];
    const float* c3_w = (const float*)d_in[5];
    const float* c3_b = (const float*)d_in[6];
    const float* p4_w = (const float*)d_in[7];
    const float* p4_b = (const float*)d_in[8];

    const int Btot   = in_sizes[0] / 784;          // 32768
    const int blocks = (Btot + 31) / 32;           // 32 images per block

    lenet1_fused<<<blocks, 256, 0, stream>>>(
        x, h1_w, h1_b, p2_w, p2_b, c3_w, c3_b, p4_w, p4_b,
        (float*)d_out, Btot);
}
